// Mamba2EnhancedBlock_22806276342454
// MI455X (gfx1250) — compile-verified, matrix operand order checked by reading, not run on hardware
//
#include <hip/hip_runtime.h>

typedef __bf16 bf16_t;
typedef __attribute__((ext_vector_type(16))) __bf16 v16bf;
typedef __attribute__((ext_vector_type(8)))  float  v8f;

#define L_SEQ    2048
#define DMODEL   768
#define DSTATE   64
#define DCONV    4
#define DINNER   1536
#define HEADDIM  64
#define NHEADS   24
#define CONVDIM  1664   // DINNER + 2*DSTATE
#define DPROJ    3224   // 2*DINNER + 2*DSTATE + NHEADS
#define HIDDEN_  1536
#define EPS_     1e-5f

__device__ __forceinline__ float silu_f(float x) { return x / (1.0f + __expf(-x)); }

// ---------------------------------------------------------------------------
// CDNA5 async global->LDS copy (one-hop, tracked by ASYNCcnt).
// GV mode: 64-bit global address in a VGPR pair, LDS byte address in VDST.
// Generic 'shared' pointers carry the LDS offset in their low 32 bits
// (ISA 10.2: LDS flat addr = addr[31:0]).
// ---------------------------------------------------------------------------
__device__ __forceinline__ void async_copy_b128(void* lds_ptr, const void* gptr)
{
    uint32_t lds_addr = (uint32_t)(uintptr_t)lds_ptr;
    uint64_t gaddr    = (uint64_t)(uintptr_t)gptr;
    asm volatile("global_load_async_to_lds_b128 %0, %1, off"
                 :: "v"(lds_addr), "v"(gaddr)
                 : "memory");
}
__device__ __forceinline__ void wait_async0()
{
    asm volatile("s_wait_asynccnt 0" ::: "memory");
}

// ---------------------------------------------------------------------------
// bf16 WMMA GEMM:  C[M,N] = A[M,K] * B[N,K]^T  (+ optional residual[M,N])
// 128x128 block tile, BK=32, 256 threads = 8 waves, each wave owns 32x64
// (2x4 tiles of 16x16), f32 accum via v_wmma_f32_16x16x32_bf16.
// Staging uses global_load_async_to_lds_b128 double buffering.
// ---------------------------------------------------------------------------
#define TM 128
#define TN 128
#define TK 32

__global__ __launch_bounds__(256) void gemm_bf16_wmma(
    const bf16_t* __restrict__ A, const bf16_t* __restrict__ Bw,
    const float* __restrict__ residual, float* __restrict__ C,
    int M, int N, int K)
{
    __shared__ __align__(16) bf16_t sA[2][TM * TK];
    __shared__ __align__(16) bf16_t sB[2][TN * TK];

    const int tid  = threadIdx.x;
    const int lane = tid & 31;
    const int wave = tid >> 5;
    const int wm = wave & 3;        // 0..3 : 32-row strip
    const int wn = wave >> 2;       // 0..1 : 64-col strip
    const int bm = blockIdx.y * TM;
    const int bn = blockIdx.x * TN;

    const int fr    = lane & 15;          // row (A) / col (B) within 16
    const int khalf = (lane >> 4) << 3;   // lanes 16-31 hold K offset +8
    const int rhi   = (lane >> 4) << 3;   // C rows: lanes 16-31 hold M+8

    v8f acc[2][4];
#pragma unroll
    for (int i = 0; i < 2; ++i)
#pragma unroll
        for (int j = 0; j < 4; ++j)
#pragma unroll
            for (int e = 0; e < 8; ++e) acc[i][j][e] = 0.0f;

    // Each thread issues 2x (A) + 2x (B) async 16B copies per K-step:
    // 512 chunks of 8 bf16 per tile, 256 threads.
    auto stage_async = [&](int buf, int k0) {
#pragma unroll
        for (int p = 0; p < 2; ++p) {
            int idx = tid + p * 256;        // 0..511
            int row = idx >> 2;             // 0..127
            int c8  = (idx & 3) << 3;       // 0,8,16,24
            async_copy_b128(&sA[buf][row * TK + c8],
                            A + (size_t)(bm + row) * K + k0 + c8);
            int brow = bn + row;
            if (brow < N)                   // edge columns: never stored, LDS
                async_copy_b128(&sB[buf][row * TK + c8],   // garbage harmless
                                Bw + (size_t)brow * K + k0 + c8);
        }
    };

    stage_async(0, 0);
    wait_async0();
    __syncthreads();

    const int nk = K / TK;
    for (int kt = 0; kt < nk; ++kt) {
        const int buf = kt & 1;
        if (kt + 1 < nk) stage_async(buf ^ 1, (kt + 1) * TK);

        union Frag { v16bf v; uint4 q[2]; };
        Frag af[2], bfr[4];
#pragma unroll
        for (int mt = 0; mt < 2; ++mt) {
            const bf16_t* base = &sA[buf][(wm * 32 + mt * 16 + fr) * TK];
            af[mt].q[0] = *reinterpret_cast<const uint4*>(base + khalf);
            af[mt].q[1] = *reinterpret_cast<const uint4*>(base + 16 + khalf);
        }
#pragma unroll
        for (int nt = 0; nt < 4; ++nt) {
            const bf16_t* base = &sB[buf][(wn * 64 + nt * 16 + fr) * TK];
            bfr[nt].q[0] = *reinterpret_cast<const uint4*>(base + khalf);
            bfr[nt].q[1] = *reinterpret_cast<const uint4*>(base + 16 + khalf);
        }
#pragma unroll
        for (int mt = 0; mt < 2; ++mt)
#pragma unroll
            for (int nt = 0; nt < 4; ++nt)
                acc[mt][nt] = __builtin_amdgcn_wmma_f32_16x16x32_bf16(
                    false, af[mt].v, false, bfr[nt].v,
                    (short)0, acc[mt][nt], false, false);

        wait_async0();      // next tile's copies landed (no-op on last iter)
        __syncthreads();
    }

#pragma unroll
    for (int mt = 0; mt < 2; ++mt) {
#pragma unroll
        for (int nt = 0; nt < 4; ++nt) {
            int col = bn + wn * 64 + nt * 16 + fr;
            if (col < N) {
#pragma unroll
                for (int e = 0; e < 8; ++e) {
                    int row = bm + wm * 32 + mt * 16 + rhi + e;
                    size_t off = (size_t)row * N + col;
                    float v = acc[mt][nt][e];
                    if (residual) v += residual[off];
                    C[off] = v;
                }
            }
        }
    }
}

// ---------------------------------------------------------------------------
// LayerNorm over D, output bf16 (feeds a WMMA GEMM)
// ---------------------------------------------------------------------------
__global__ __launch_bounds__(256) void layernorm_to_bf16(
    const float* __restrict__ x, const float* __restrict__ w,
    const float* __restrict__ b, bf16_t* __restrict__ out, int D)
{
    const int row = blockIdx.x;
    const int tid = threadIdx.x;
    const float* xr = x + (size_t)row * D;
    __shared__ float red[256];

    float s = 0.f;
    for (int i = tid; i < D; i += 256) s += xr[i];
    red[tid] = s; __syncthreads();
    for (int o = 128; o > 0; o >>= 1) { if (tid < o) red[tid] += red[tid + o]; __syncthreads(); }
    const float mean = red[0] / (float)D;
    __syncthreads();
    float v = 0.f;
    for (int i = tid; i < D; i += 256) { float d = xr[i] - mean; v += d * d; }
    red[tid] = v; __syncthreads();
    for (int o = 128; o > 0; o >>= 1) { if (tid < o) red[tid] += red[tid + o]; __syncthreads(); }
    const float rstd = rsqrtf(red[0] / (float)D + EPS_);
    for (int i = tid; i < D; i += 256)
        out[(size_t)row * D + i] = (bf16_t)((xr[i] - mean) * rstd * w[i] + b[i]);
}

// ---------------------------------------------------------------------------
// Depthwise causal conv (width 4) + bias + SiLU over xBC channels
// ---------------------------------------------------------------------------
__global__ __launch_bounds__(256) void conv_silu_kernel(
    const float* __restrict__ zxbcdt, const float* __restrict__ conv_w,
    const float* __restrict__ conv_b, float* __restrict__ xBC)
{
    int idx = blockIdx.x * 256 + threadIdx.x;
    if (idx >= L_SEQ * CONVDIM) return;
    int t = idx / CONVDIM;
    int c = idx - t * CONVDIM;
    float acc = conv_b[c];
#pragma unroll
    for (int k = 0; k < DCONV; ++k) {
        int tt = t - (DCONV - 1) + k;
        if (tt >= 0)
            acc += zxbcdt[(size_t)tt * DPROJ + DINNER + c] * conv_w[c * DCONV + k];
    }
    xBC[idx] = silu_f(acc);
}

// ---------------------------------------------------------------------------
// dt = softplus(dt_raw + bias); dA = exp(-exp(A_log)*dt)
// ---------------------------------------------------------------------------
__global__ __launch_bounds__(256) void dt_kernel(
    const float* __restrict__ zxbcdt, const float* __restrict__ dt_bias,
    const float* __restrict__ A_log, float* __restrict__ dt_out,
    float* __restrict__ dA_out)
{
    int idx = blockIdx.x * 256 + threadIdx.x;
    if (idx >= L_SEQ * NHEADS) return;
    int t = idx / NHEADS;
    int h = idx - t * NHEADS;
    float raw = zxbcdt[(size_t)t * DPROJ + DINNER + CONVDIM + h] + dt_bias[h];
    float dtv = (raw > 20.f) ? raw : log1pf(__expf(raw));
    dt_out[idx] = dtv;
    dA_out[idx] = __expf(-__expf(A_log[h]) * dtv);
}

// ---------------------------------------------------------------------------
// Sequential SSM scan. One block per head; 64x64 state register-resident
// (16 f32 per lane). B/C/x rows staged in LDS each step; y reduced with
// wave32 shuffles.
// ---------------------------------------------------------------------------
__global__ __launch_bounds__(256) void ssm_scan_kernel(
    const float* __restrict__ xBC, const float* __restrict__ dt,
    const float* __restrict__ dA, float* __restrict__ ys)
{
    const int head = blockIdx.x;
    const int tid = threadIdx.x;
    const int p  = tid >> 2;           // head-dim row 0..63
    const int s0 = (tid & 3) << 4;     // 16-state slice
    float st[16];
#pragma unroll
    for (int i = 0; i < 16; ++i) st[i] = 0.f;
    __shared__ float sB[DSTATE], sC[DSTATE], sX[HEADDIM];

    for (int t = 0; t < L_SEQ; ++t) {
        const float* row = xBC + (size_t)t * CONVDIM;
        if (tid < 64)       sB[tid]       = row[DINNER + tid];
        else if (tid < 128) sC[tid - 64]  = row[DINNER + DSTATE + (tid - 64)];
        else if (tid < 192) sX[tid - 128] = row[head * HEADDIM + (tid - 128)];
        __syncthreads();
        const float dAv = dA[t * NHEADS + head];
        const float dtx = dt[t * NHEADS + head] * sX[p];
        float acc = 0.f;
#pragma unroll
        for (int i = 0; i < 16; ++i) {
            float h2 = st[i] * dAv + dtx * sB[s0 + i];
            st[i] = h2;
            acc += h2 * sC[s0 + i];
        }
        acc += __shfl_xor(acc, 1, 32);
        acc += __shfl_xor(acc, 2, 32);
        if ((tid & 3) == 0)
            ys[(size_t)t * DINNER + head * HEADDIM + p] = acc;
        __syncthreads();
    }
}

// ---------------------------------------------------------------------------
// y = (ys + D*xs) * silu(z); RMSNorm(1536) * w ; output bf16
// ---------------------------------------------------------------------------
__global__ __launch_bounds__(256) void gate_rmsnorm_kernel(
    const float* __restrict__ zxbcdt, const float* __restrict__ xBC,
    const float* __restrict__ ys, const float* __restrict__ Dskip,
    const float* __restrict__ nw, bf16_t* __restrict__ out)
{
    const int t = blockIdx.x;
    const int tid = threadIdx.x;
    __shared__ float red[256];
    __shared__ float g[DINNER];
    float ss = 0.f;
    for (int i = tid; i < DINNER; i += 256) {
        float xs = xBC[(size_t)t * CONVDIM + i];
        float y  = ys[(size_t)t * DINNER + i] + Dskip[i >> 6] * xs;
        float z  = zxbcdt[(size_t)t * DPROJ + i];
        float gv = y * silu_f(z);
        g[i] = gv;
        ss += gv * gv;
    }
    red[tid] = ss; __syncthreads();
    for (int o = 128; o > 0; o >>= 1) { if (tid < o) red[tid] += red[tid + o]; __syncthreads(); }
    const float r = rsqrtf(red[0] / (float)DINNER + EPS_);
    for (int i = tid; i < DINNER; i += 256)
        out[(size_t)t * DINNER + i] = (bf16_t)(g[i] * r * nw[i]);
}

// ---------------------------------------------------------------------------
// SwiGLU gate: g = f[:, :H] * silu(f[:, H:]) ; output bf16
// ---------------------------------------------------------------------------
__global__ __launch_bounds__(256) void glu_kernel(
    const float* __restrict__ f, bf16_t* __restrict__ out)
{
    int idx = blockIdx.x * 256 + threadIdx.x;
    if (idx >= L_SEQ * HIDDEN_) return;
    int t = idx / HIDDEN_;
    int i = idx - t * HIDDEN_;
    float a = f[(size_t)t * (2 * HIDDEN_) + i];
    float b = f[(size_t)t * (2 * HIDDEN_) + HIDDEN_ + i];
    out[idx] = (bf16_t)(a * silu_f(b));
}

// ---------------------------------------------------------------------------
// f32 -> bf16 conversion (weights, once per launch — deterministic)
// ---------------------------------------------------------------------------
__global__ __launch_bounds__(256) void f2bf_kernel(
    const float* __restrict__ in, bf16_t* __restrict__ out, int n)
{
    int idx = blockIdx.x * 256 + threadIdx.x;
    if (idx < n) out[idx] = (bf16_t)in[idx];
}

// ---------------------------------------------------------------------------
extern "C" void kernel_launch(void* const* d_in, const int* in_sizes, int n_in,
                              void* d_out, int out_size, void* d_ws, size_t ws_size,
                              hipStream_t stream)
{
    const float* x        = (const float*)d_in[0];
    const float* ln1_w    = (const float*)d_in[1];
    const float* ln1_b    = (const float*)d_in[2];
    const float* ln2_w    = (const float*)d_in[3];
    const float* ln2_b    = (const float*)d_in[4];
    const float* in_proj  = (const float*)d_in[5];
    const float* conv_w   = (const float*)d_in[6];
    const float* conv_b   = (const float*)d_in[7];
    const float* dt_bias  = (const float*)d_in[8];
    const float* A_log    = (const float*)d_in[9];
    const float* D_skip   = (const float*)d_in[10];
    const float* ssm_nw   = (const float*)d_in[11];
    const float* out_proj = (const float*)d_in[12];
    const float* fc1      = (const float*)d_in[13];
    const float* fc2      = (const float*)d_in[14];
    float* out = (float*)d_out;

    char* wsp = (char*)d_ws;
    size_t off = 0;
    auto alloc = [&](size_t bytes) -> void* {
        off = (off + 255) & ~(size_t)255;
        void* p = wsp + off;
        off += bytes;
        return p;
    };

    bf16_t* w_in_bf  = (bf16_t*)alloc((size_t)DPROJ * DMODEL * 2);
    bf16_t* w_out_bf = (bf16_t*)alloc((size_t)DMODEL * DINNER * 2);
    bf16_t* w_fc1_bf = (bf16_t*)alloc((size_t)(2 * HIDDEN_) * DMODEL * 2);
    bf16_t* w_fc2_bf = (bf16_t*)alloc((size_t)DMODEL * HIDDEN_ * 2);
    bf16_t* h1_bf    = (bf16_t*)alloc((size_t)L_SEQ * DMODEL * 2);
    float*  zxbcdt   = (float*) alloc((size_t)L_SEQ * DPROJ * 4);
    float*  xBCc     = (float*) alloc((size_t)L_SEQ * CONVDIM * 4);
    float*  dtb      = (float*) alloc((size_t)L_SEQ * NHEADS * 4);
    float*  dAb      = (float*) alloc((size_t)L_SEQ * NHEADS * 4);
    float*  ysb      = (float*) alloc((size_t)L_SEQ * DINNER * 4);
    bf16_t* y_bf     = (bf16_t*)alloc((size_t)L_SEQ * DINNER * 2);
    float*  x2       = (float*) alloc((size_t)L_SEQ * DMODEL * 4);
    bf16_t* h2_bf    = (bf16_t*)alloc((size_t)L_SEQ * DMODEL * 2);
    float*  fbuf     = (float*) alloc((size_t)L_SEQ * 2 * HIDDEN_ * 4);
    bf16_t* g_bf     = (bf16_t*)alloc((size_t)L_SEQ * HIDDEN_ * 2);

    auto cvt = [&](const float* src, bf16_t* dst, int n) {
        f2bf_kernel<<<(n + 255) / 256, 256, 0, stream>>>(src, dst, n);
    };
    cvt(in_proj,  w_in_bf,  DPROJ * DMODEL);
    cvt(out_proj, w_out_bf, DMODEL * DINNER);
    cvt(fc1,      w_fc1_bf, 2 * HIDDEN_ * DMODEL);
    cvt(fc2,      w_fc2_bf, DMODEL * HIDDEN_);

    // LN1 -> bf16
    layernorm_to_bf16<<<L_SEQ, 256, 0, stream>>>(x, ln1_w, ln1_b, h1_bf, DMODEL);

    // in_proj: [2048,3224] = h1 @ W_in^T
    dim3 g1((DPROJ + TN - 1) / TN, L_SEQ / TM);
    gemm_bf16_wmma<<<g1, 256, 0, stream>>>(h1_bf, w_in_bf, nullptr, zxbcdt,
                                           L_SEQ, DPROJ, DMODEL);

    conv_silu_kernel<<<(L_SEQ * CONVDIM + 255) / 256, 256, 0, stream>>>(
        zxbcdt, conv_w, conv_b, xBCc);
    dt_kernel<<<(L_SEQ * NHEADS + 255) / 256, 256, 0, stream>>>(
        zxbcdt, dt_bias, A_log, dtb, dAb);
    ssm_scan_kernel<<<NHEADS, 256, 0, stream>>>(xBCc, dtb, dAb, ysb);
    gate_rmsnorm_kernel<<<L_SEQ, 256, 0, stream>>>(
        zxbcdt, xBCc, ysb, D_skip, ssm_nw, y_bf);

    // out_proj + residual(x) -> x2
    dim3 g2(DMODEL / TN, L_SEQ / TM);
    gemm_bf16_wmma<<<g2, 256, 0, stream>>>(y_bf, w_out_bf, x, x2,
                                           L_SEQ, DMODEL, DINNER);

    // LN2 -> bf16
    layernorm_to_bf16<<<L_SEQ, 256, 0, stream>>>(x2, ln2_w, ln2_b, h2_bf, DMODEL);

    // fc1
    dim3 g3((2 * HIDDEN_) / TN, L_SEQ / TM);
    gemm_bf16_wmma<<<g3, 256, 0, stream>>>(h2_bf, w_fc1_bf, nullptr, fbuf,
                                           L_SEQ, 2 * HIDDEN_, DMODEL);

    glu_kernel<<<(L_SEQ * HIDDEN_ + 255) / 256, 256, 0, stream>>>(fbuf, g_bf);

    // fc2 + residual(x2) -> out
    gemm_bf16_wmma<<<g2, 256, 0, stream>>>(g_bf, w_fc2_bf, x2, out,
                                           L_SEQ, DMODEL, HIDDEN_);

    (void)in_sizes; (void)n_in; (void)out_size; (void)ws_size;
}
